// ArtOutBlock_35777077576021
// MI455X (gfx1250) — matrix operand order repaired and checked by reading, not compiled, not run on hardware
//
#include <hip/hip_runtime.h>
#include <hip/hip_bf16.h>

// ---------------- problem constants (match reference) ----------------
#define NB    16
#define NP    1024
#define NA    12
#define NROT  60
#define C0c   64
#define C1c   128
#define CINc  256
#define CFEATc 512
#define COUTc 256
#define PCH   32        // p-chunks per batch for pointnet partial-max
#define EPSc  1e-5f
#define NEGBIG (-3.0e38f)

typedef __attribute__((ext_vector_type(16))) __bf16 v16bf;
typedef __attribute__((ext_vector_type(8)))  float  v8f;

// WMMA 16x16x32 bf16 lane layouts (wave32, §7.12.2):
//  A (16x32 MxK): M = lane&15, half = lane>>4, elem j -> K = (j>>3)*16 + half*8 + (j&7)
//  B (32x16 KxN): N = lane&15; this code loads A and B with elem j -> K = half*16 + j
//  D (16x16 f32): vgpr r -> M = r + half*8, N = lane&15

// A-fragment from row-major W (ld = K, K % 32 == 0 -> 32B-aligned chunks):
// two float4 pairs per fragment guarantee global_load_b128.
__device__ __forceinline__ v16bf a_frag_vec(const float* __restrict__ W, long row_off,
                                            int kbase, int half) {
  const float4* wp = (const float4*)(W + row_off + kbase + (half << 4));
  float4 w0 = wp[0], w1 = wp[1];   // k = base .. base+7   (elems 0..7)
  float4 w2 = wp[2], w3 = wp[3];   // k = base+8..base+15  (elems 8..15)
  v16bf a;
  a[0] = (__bf16)w0.x;  a[1] = (__bf16)w0.y;  a[2] = (__bf16)w0.z;  a[3] = (__bf16)w0.w;
  a[4] = (__bf16)w1.x;  a[5] = (__bf16)w1.y;  a[6] = (__bf16)w1.z;  a[7] = (__bf16)w1.w;
  a[8] = (__bf16)w2.x;  a[9] = (__bf16)w2.y;  a[10] = (__bf16)w2.z; a[11] = (__bf16)w2.w;
  a[12] = (__bf16)w3.x; a[13] = (__bf16)w3.y; a[14] = (__bf16)w3.z; a[15] = (__bf16)w3.w;
  return a;
}

// ======================================================================
// Kernel 1: pointnet GEMM with fused max over points (C compile-time).
//   feats part (K = C, multiple of 32) via WMMA, branch-free loads;
//   xyz rank-3 part applied as VALU FMAs on the accumulator.
// grid.x = NB*PCH, grid.y = C/32 (two M-tiles per wave), block = 32.
// ======================================================================
template <int C>
__global__ void __launch_bounds__(32)
pointnet_max_wmma(const float* __restrict__ W, const float* __restrict__ xyz,
                  const float* __restrict__ feats, float* __restrict__ part) {
  const int lane = threadIdx.x;
  const int half = lane >> 4;
  const int nidx = lane & 15;
  const int aL   = (nidx < NA) ? nidx : (NA - 1);   // clamped: safe, discarded cols
  const int mt0  = blockIdx.y * 2;
  const int b    = blockIdx.x / PCH;
  const int pc   = blockIdx.x % PCH;
  constexpr int PP = NP / PCH;
  constexpr int KT = C + 3;
  constexpr int KS = C / 32;                        // 2 (C=64) or 4 (C=128)

  // A fragments: W feats columns (col = k + 3), hoisted out of the p loop.
  // (ld = C+3 is odd -> scalar loads, but this runs once per wave.)
  v16bf afr[2][KS];
#pragma unroll
  for (int mi = 0; mi < 2; ++mi) {
    const int m = (mt0 + mi) * 16 + nidx;
#pragma unroll
    for (int ks = 0; ks < KS; ++ks) {
      v16bf a;
#pragma unroll
      for (int j = 0; j < 16; ++j) {
        int k = (ks << 5) + (half << 4) + j;
        a[j] = (__bf16)W[m * KT + 3 + k];
      }
      afr[mi][ks] = a;
    }
  }
  // xyz weight columns for this lane's output rows.
  float Wx[2][8][3];
#pragma unroll
  for (int mi = 0; mi < 2; ++mi)
#pragma unroll
    for (int r = 0; r < 8; ++r)
#pragma unroll
      for (int t = 0; t < 3; ++t)
        Wx[mi][r][t] = W[((mt0 + mi) * 16 + half * 8 + r) * KT + t];

  v8f rm0, rm1;
#pragma unroll
  for (int r = 0; r < 8; ++r) { rm0[r] = NEGBIG; rm1[r] = NEGBIG; }

  constexpr long kstr = (long)NP * NA;              // feats channel stride (floats)
  for (int pi = 0; pi < PP; ++pi) {
    const int p = pc * PP + pi;
    const float x0 = xyz[(b * 3 + 0) * NP + p];
    const float x1 = xyz[(b * 3 + 1) * NP + p];
    const float x2 = xyz[(b * 3 + 2) * NP + p];
    v8f acc0 = {}, acc1 = {};
#pragma unroll
    for (int ks = 0; ks < KS; ++ks) {
      const float* bp =
          feats + ((long)(b * C + (ks << 5) + (half << 4)) * NP + p) * NA + aL;
      v16bf bf;
#pragma unroll
      for (int j = 0; j < 16; ++j) bf[j] = (__bf16)bp[(long)j * kstr];
      acc0 = __builtin_amdgcn_wmma_f32_16x16x32_bf16(false, afr[0][ks], false, bf,
                                                     (short)0, acc0, false, false);
      acc1 = __builtin_amdgcn_wmma_f32_16x16x32_bf16(false, afr[1][ks], false, bf,
                                                     (short)0, acc1, false, false);
    }
#pragma unroll
    for (int r = 0; r < 8; ++r) {
      rm0[r] = fmaxf(rm0[r], acc0[r] + Wx[0][r][0] * x0 + Wx[0][r][1] * x1 + Wx[0][r][2] * x2);
      rm1[r] = fmaxf(rm1[r], acc1[r] + Wx[1][r][0] * x0 + Wx[1][r][1] * x1 + Wx[1][r][2] * x2);
    }
  }
  if (nidx < NA) {
#pragma unroll
    for (int r = 0; r < 8; ++r) {
      int m0 = mt0 * 16 + half * 8 + r;
      part[((long)(b * PCH + pc) * C + m0) * NA + nidx]        = rm0[r];
      part[((long)(b * PCH + pc) * C + (m0 + 16)) * NA + nidx] = rm1[r];
    }
  }
}

// Kernel 2: reduce partial maxes over PCH chunks, add bias.
__global__ void maxreduce_bias(const float* __restrict__ part,
                               const float* __restrict__ bias,
                               float* __restrict__ ymax, int C) {
  int t = blockIdx.x * blockDim.x + threadIdx.x;
  int total = NB * C * NA;
  if (t >= total) return;
  int a = t % NA, m = (t / NA) % C, b = t / (NA * C);
  float v = NEGBIG;
  for (int pc = 0; pc < PCH; ++pc)
    v = fmaxf(v, part[((long)(b * PCH + pc) * C + m) * NA + a]);
  ymax[t] = v + bias[m];
}

// Kernel 3: batchnorm with on-the-fly stats over (b, a) per channel m.
__global__ void bn_stats_apply(const float* __restrict__ Y,
                               const float* __restrict__ g, const float* __restrict__ bb,
                               float* __restrict__ dst, int B, int M, int A,
                               long db, long dm, long da, int coff, int relu) {
  const int m = blockIdx.x;
  const int n = B * A;
  __shared__ float s1[256], s2[256];
  float sum = 0.f, sq = 0.f;
  for (int e = threadIdx.x; e < n; e += blockDim.x) {
    int b = e / A, a = e - b * A;
    float v = Y[((long)b * M + m) * A + a];
    sum += v; sq += v * v;
  }
  s1[threadIdx.x] = sum; s2[threadIdx.x] = sq;
  __syncthreads();
  for (int off = 128; off > 0; off >>= 1) {
    if ((int)threadIdx.x < off) {
      s1[threadIdx.x] += s1[threadIdx.x + off];
      s2[threadIdx.x] += s2[threadIdx.x + off];
    }
    __syncthreads();
  }
  const float mean = s1[0] / (float)n;
  const float var  = s2[0] / (float)n - mean * mean;
  const float rstd = rsqrtf(var + EPSc);
  const float gg = g[m], bv = bb[m];
  for (int e = threadIdx.x; e < n; e += blockDim.x) {
    int b = e / A, a = e - b * A;
    float v = (Y[((long)b * M + m) * A + a] - mean) * rstd * gg + bv;
    if (relu) v = fmaxf(v, 0.f);
    dst[(long)b * db + (long)(m + coff) * dm + (long)a * da] = v;
  }
}

// ======================================================================
// Kernel 4: generic WMMA GEMM, all strides compile-time.
//   Y(b,m,a) = sum_k W[m,k]*X[b*SB+k*SK+a*SA] + bias[m]
//   Preconditions: K % 32 == 0, M % 16 == 0 (true for all instantiations).
// grid.x = B*ceil(A/16), grid.y = M/16, block = 32.
// ======================================================================
template <int A, int SB, int SK, int SA, int YB, int YM, int YA>
__global__ void __launch_bounds__(32)
gemm_bias_wmma(const float* __restrict__ W, const float* __restrict__ bias,
               const float* __restrict__ X, float* __restrict__ Y, int M, int K) {
  const int lane = threadIdx.x;
  const int half = lane >> 4;
  const int nsub = lane & 15;
  constexpr int NT = (A + 15) / 16;
  const int b  = blockIdx.x / NT;
  const int nt = blockIdx.x % NT;
  const int mt = blockIdx.y;
  const int aidx = nt * 16 + nsub;
  const int aL   = (aidx < A) ? aidx : (A - 1);   // clamp loads (cols discarded)
  const long wrow = (long)(mt * 16 + nsub) * K;
  v8f acc = {};
  const int ksteps = K >> 5;
  for (int ks = 0; ks < ksteps; ++ks) {
    const int kbase = ks << 5;
    v16bf af = a_frag_vec(W, wrow, kbase, half);
    const float* xp = X + (long)b * SB + (long)(kbase + (half << 4)) * SK + (long)aL * SA;
    v16bf bf;
#pragma unroll
    for (int j = 0; j < 16; ++j) bf[j] = (__bf16)xp[(long)j * SK];
    acc = __builtin_amdgcn_wmma_f32_16x16x32_bf16(false, af, false, bf,
                                                  (short)0, acc, false, false);
  }
  if (aidx < A) {
#pragma unroll
    for (int r = 0; r < 8; ++r) {
      int m = mt * 16 + half * 8 + r;
      Y[(long)b * YB + (long)m * YM + (long)aidx * YA] = acc[r] + bias[m];
    }
  }
}

// ======================================================================
// Kernel 5: feat GEMM fused with trace_idx gather (branch-free B frag).
//   xg[b, c*12+jj, r] = xlin[b, c, trace[jj,r]]
// grid.x = NB*4, grid.y = 32, block = 32.
// ======================================================================
__global__ void __launch_bounds__(32)
feat_gather_gemm_wmma(const float* __restrict__ W, const float* __restrict__ bias,
                      const float* __restrict__ xlin, const int* __restrict__ trace,
                      float* __restrict__ Y) {
  const int lane = threadIdx.x;
  const int half = lane >> 4;
  const int nsub = lane & 15;
  const int b  = blockIdx.x >> 2;
  const int nt = blockIdx.x & 3;
  const int mt = blockIdx.y;
  const int r  = nt * 16 + nsub;
  const int rL = (r < NROT) ? r : (NROT - 1);
  constexpr int K = CINc * NA;                     // 3072
  const long wrow = (long)(mt * 16 + nsub) * K;
  v8f acc = {};
  for (int ks = 0; ks < (K >> 5); ++ks) {
    const int kbase = ks << 5;
    v16bf af = a_frag_vec(W, wrow, kbase, half);
    v16bf bf;
#pragma unroll
    for (int j = 0; j < 16; ++j) {
      int k = kbase + (half << 4) + j;
      int c = k / NA;
      int jj = k - c * NA;
      int asrc = trace[jj * NROT + rL];
      bf[j] = (__bf16)xlin[((long)b * CINc + c) * NA + asrc];
    }
    acc = __builtin_amdgcn_wmma_f32_16x16x32_bf16(false, af, false, bf,
                                                  (short)0, acc, false, false);
  }
  if (r < NROT) {
#pragma unroll
    for (int r8 = 0; r8 < 8; ++r8) {
      int m = mt * 16 + half * 8 + r8;
      Y[((long)b * CFEATc + m) * NROT + r] = acc[r8] + bias[m];
    }
  }
}

// Kernel 6: reg_W2 / att_W2 heads (tiny GEMVs), one thread per (b, r).
__global__ void heads_kernel(const float* __restrict__ h, const float* __restrict__ aatt,
                             const float* __restrict__ rW2, const float* __restrict__ rb2,
                             const float* __restrict__ aW2, const float* __restrict__ ab2,
                             float* __restrict__ resid, float* __restrict__ logits) {
  int t = blockIdx.x * blockDim.x + threadIdx.x;
  if (t >= NB * NROT) return;
  int b = t / NROT, r = t % NROT;
  for (int o = 0; o < 7; ++o) {
    float s = rb2[o];
    for (int c = 0; c < 128; ++c) s += rW2[o * 128 + c] * h[((long)b * 128 + c) * NROT + r];
    resid[t * 7 + o] = s;
  }
  float s = ab2[0];
  for (int c = 0; c < 128; ++c) s += aW2[c] * aatt[((long)b * 128 + c) * NROT + r];
  logits[t] = s;
}

// Kernel 7: residual -> exp_so3 -> anchors @ R, and res_T passthrough.
__global__ void pose_kernel(const float* __restrict__ resid,
                            const float* __restrict__ anchors,
                            float* __restrict__ predR, float* __restrict__ resT) {
  int t = blockIdx.x * blockDim.x + threadIdx.x;
  if (t >= NB * NROT) return;
  int r = t % NROT;
  const float* rd = resid + t * 7;
  float dx = rd[0], dy = rd[1], dz = rd[2];
  float nrm = sqrtf(dx * dx + dy * dy + dz * dz);
  dx /= nrm; dy /= nrm; dz /= nrm;
  float sig = 1.0f / (1.0f + expf(-rd[3]));
  float ang = (sig - 0.5f) * 3.14159265358979323846f / 5.0f;
  float vx = dx * ang, vy = dy * ang, vz = dz * ang;
  float th = sqrtf(vx * vx + vy * vy + vz * vz);
  float inv = (th < 1e-8f) ? 1.0f : 1.0f / th;
  float kx = vx * inv, ky = vy * inv, kz = vz * inv;
  float s = sinf(th), c = 1.0f - cosf(th);
  float Km[9] = {0.f, -kz, ky, kz, 0.f, -kx, -ky, kx, 0.f};
  float K2[9];
#pragma unroll
  for (int i = 0; i < 3; ++i)
#pragma unroll
    for (int j = 0; j < 3; ++j) {
      float a = 0.f;
#pragma unroll
      for (int kk = 0; kk < 3; ++kk) a += Km[i * 3 + kk] * Km[kk * 3 + j];
      K2[i * 3 + j] = a;
    }
  float R[9];
#pragma unroll
  for (int i = 0; i < 9; ++i)
    R[i] = ((i % 4) == 0 ? 1.0f : 0.0f) + s * Km[i] + c * K2[i];
  const float* An = anchors + r * 9;
#pragma unroll
  for (int i = 0; i < 3; ++i)
#pragma unroll
    for (int j = 0; j < 3; ++j) {
      float a = 0.f;
#pragma unroll
      for (int kk = 0; kk < 3; ++kk) a += An[i * 3 + kk] * R[kk * 3 + j];
      predR[t * 9 + i * 3 + j] = a;
    }
  resT[t * 3 + 0] = rd[4];
  resT[t * 3 + 1] = rd[5];
  resT[t * 3 + 2] = rd[6];
}

// Kernel 8: softmax over the 60 rotations, one block per batch.
__global__ void softmax_kernel(const float* __restrict__ logits, float* __restrict__ out) {
  int b = blockIdx.x;
  int t = threadIdx.x;
  __shared__ float sh[64];
  float v = (t < NROT) ? logits[b * NROT + t] : NEGBIG;
  sh[t] = v;
  __syncthreads();
  for (int off = 32; off > 0; off >>= 1) {
    if (t < off) sh[t] = fmaxf(sh[t], sh[t + off]);
    __syncthreads();
  }
  float mx = sh[0];
  __syncthreads();
  float e = (t < NROT) ? expf(v - mx) : 0.0f;
  sh[t] = e;
  __syncthreads();
  for (int off = 32; off > 0; off >>= 1) {
    if (t < off) sh[t] += sh[t + off];
    __syncthreads();
  }
  float sum = sh[0];
  if (t < NROT) out[b * NROT + t] = e / sum;
}

// Kernel 9: max over rotations -> xm[b*512+m]
__global__ void rowmax_kernel(const float* __restrict__ xf, float* __restrict__ xm) {
  int t = blockIdx.x * blockDim.x + threadIdx.x;
  if (t >= NB * CFEATc) return;
  const float* p = xf + (long)t * NROT;
  float v = NEGBIG;
  for (int r = 0; r < NROT; ++r) v = fmaxf(v, p[r]);
  xm[t] = v;
}

// ======================================================================
extern "C" void kernel_launch(void* const* d_in, const int* in_sizes, int n_in,
                              void* d_out, int out_size, void* d_ws, size_t ws_size,
                              hipStream_t stream) {
  (void)in_sizes; (void)n_in; (void)out_size; (void)ws_size;
  const float* xyz0   = (const float*)d_in[0];
  const float* feats0 = (const float*)d_in[1];
  const float* xyz1   = (const float*)d_in[2];
  const float* feats1 = (const float*)d_in[3];
  const int*   trace  = (const int*)  d_in[4];
  const float* anchors= (const float*)d_in[5];
  const float* pn0_W  = (const float*)d_in[6];
  const float* pn0_b  = (const float*)d_in[7];
  const float* pn0_g  = (const float*)d_in[8];
  const float* pn0_bb = (const float*)d_in[9];
  const float* pn1_W  = (const float*)d_in[10];
  const float* pn1_b  = (const float*)d_in[11];
  const float* pn1_g  = (const float*)d_in[12];
  const float* pn1_bb = (const float*)d_in[13];
  const float* lin_W  = (const float*)d_in[14];
  const float* lin_b  = (const float*)d_in[15];
  const float* lin_g  = (const float*)d_in[16];
  const float* lin_bb = (const float*)d_in[17];
  const float* feat_W = (const float*)d_in[18];
  const float* feat_b = (const float*)d_in[19];
  const float* feat_g = (const float*)d_in[20];
  const float* feat_bb= (const float*)d_in[21];
  const float* att_W1 = (const float*)d_in[22];
  const float* att_b1 = (const float*)d_in[23];
  const float* att_g  = (const float*)d_in[24];
  const float* att_bb = (const float*)d_in[25];
  const float* att_W2 = (const float*)d_in[26];
  const float* att_b2 = (const float*)d_in[27];
  const float* reg_W1 = (const float*)d_in[28];
  const float* reg_b1 = (const float*)d_in[29];
  const float* reg_g  = (const float*)d_in[30];
  const float* reg_bb = (const float*)d_in[31];
  const float* reg_W2 = (const float*)d_in[32];
  const float* reg_b2 = (const float*)d_in[33];
  const float* out_W1 = (const float*)d_in[34];
  const float* out_b1 = (const float*)d_in[35];
  const float* out_g  = (const float*)d_in[36];
  const float* out_bb = (const float*)d_in[37];
  const float* out_W2 = (const float*)d_in[38];
  const float* out_b2 = (const float*)d_in[39];

  float* out = (float*)d_out;
  float* ws  = (float*)d_ws;

  // workspace layout (floats), total ~2.86M floats (~11.5 MB)
  float* part0  = ws + 0;         //  16*32*64*12  = 393216
  float* part1  = ws + 393216;    //  16*32*128*12 = 786432
  float* y0max  = ws + 1179648;   //  12288
  float* y1max  = ws + 1191936;   //  24576
  float* xcat   = ws + 1216512;   //  16*192*12 = 36864
  float* ylin   = ws + 1253376;   //  49152
  float* xlin   = ws + 1302528;   //  49152
  float* yfeat  = ws + 1351680;   //  16*512*60 = 491520
  float* xf     = ws + 1843200;   //  491520
  float* hraw   = ws + 2334720;   //  122880
  float* hbuf   = ws + 2457600;   //  122880
  float* araw   = ws + 2580480;   //  122880
  float* abuf   = ws + 2703360;   //  122880
  float* resid  = ws + 2826240;   //  6720
  float* logit  = ws + 2832960;   //  960
  float* xm     = ws + 2833920;   //  8192
  float* o1raw  = ws + 2842112;   //  8192
  float* obuf   = ws + 2850304;   //  8192

  // d_out: x_out[16,256] | x_attn[16,60] | pred_R[16,60,3,3] | res_T[16,60,3]
  float* out_xout = out;
  float* out_attn = out + 4096;
  float* out_predR= out + 5056;
  float* out_resT = out + 13696;

  // --- pointnet branches (WMMA + fused max) ---
  pointnet_max_wmma<C0c><<<dim3(NB * PCH, C0c / 32), 32, 0, stream>>>(pn0_W, xyz0, feats0, part0);
  pointnet_max_wmma<C1c><<<dim3(NB * PCH, C1c / 32), 32, 0, stream>>>(pn1_W, xyz1, feats1, part1);
  maxreduce_bias<<<(NB * C0c * NA + 255) / 256, 256, 0, stream>>>(part0, pn0_b, y0max, C0c);
  maxreduce_bias<<<(NB * C1c * NA + 255) / 256, 256, 0, stream>>>(part1, pn1_b, y1max, C1c);
  bn_stats_apply<<<C0c, 256, 0, stream>>>(y0max, pn0_g, pn0_bb, xcat, NB, C0c, NA,
                                          192 * NA, NA, 1, 0, 1);
  bn_stats_apply<<<C1c, 256, 0, stream>>>(y1max, pn1_g, pn1_bb, xcat, NB, C1c, NA,
                                          192 * NA, NA, 1, C0c, 1);

  // --- lin: (256,192) @ xcat (16,192,12) ---
  gemm_bias_wmma<NA, 192 * NA, NA, 1, CINc * NA, NA, 1>
      <<<dim3(NB, CINc / 16), 32, 0, stream>>>(lin_W, lin_b, xcat, ylin, CINc, 192);
  bn_stats_apply<<<CINc, 256, 0, stream>>>(ylin, lin_g, lin_bb, xlin, NB, CINc, NA,
                                           (long)CINc * NA, NA, 1, 0, 0);

  // --- feat: (512,3072) @ gathered (16,3072,60), gather fused ---
  feat_gather_gemm_wmma<<<dim3(NB * 4, CFEATc / 16), 32, 0, stream>>>(
      feat_W, feat_b, xlin, trace, yfeat);
  bn_stats_apply<<<CFEATc, 256, 0, stream>>>(yfeat, feat_g, feat_bb, xf, NB, CFEATc, NROT,
                                             (long)CFEATc * NROT, NROT, 1, 0, 1);

  // --- reg / att hidden layers: (128,512) @ xf (16,512,60) ---
  gemm_bias_wmma<NROT, CFEATc * NROT, NROT, 1, 128 * NROT, NROT, 1>
      <<<dim3(NB * 4, 128 / 16), 32, 0, stream>>>(reg_W1, reg_b1, xf, hraw, 128, CFEATc);
  bn_stats_apply<<<128, 256, 0, stream>>>(hraw, reg_g, reg_bb, hbuf, NB, 128, NROT,
                                          (long)128 * NROT, NROT, 1, 0, 1);
  gemm_bias_wmma<NROT, CFEATc * NROT, NROT, 1, 128 * NROT, NROT, 1>
      <<<dim3(NB * 4, 128 / 16), 32, 0, stream>>>(att_W1, att_b1, xf, araw, 128, CFEATc);
  bn_stats_apply<<<128, 256, 0, stream>>>(araw, att_g, att_bb, abuf, NB, 128, NROT,
                                          (long)128 * NROT, NROT, 1, 0, 1);

  // --- tiny heads, pose math, softmax ---
  heads_kernel<<<(NB * NROT + 255) / 256, 256, 0, stream>>>(
      hbuf, abuf, reg_W2, reg_b2, att_W2, att_b2, resid, logit);
  pose_kernel<<<(NB * NROT + 255) / 256, 256, 0, stream>>>(resid, anchors, out_predR, out_resT);
  softmax_kernel<<<NB, 64, 0, stream>>>(logit, out_attn);

  // --- output branch: max over rotations, two dense layers ---
  rowmax_kernel<<<(NB * CFEATc + 255) / 256, 256, 0, stream>>>(xf, xm);
  // out1: M=512,K=512, A=16(batch); X(k,a)=xm[a*512+k]; Y=[m*16+a]
  gemm_bias_wmma<NB, 0, 1, CFEATc, 0, NB, 1>
      <<<dim3(1, CFEATc / 16), 32, 0, stream>>>(out_W1, out_b1, xm, o1raw, CFEATc, CFEATc);
  bn_stats_apply<<<CFEATc, 256, 0, stream>>>(o1raw, out_g, out_bb, obuf, 1, CFEATc, NB,
                                             0, NB, 1, 0, 1);
  // out2: M=256,K=512, A=16; X(k,a)=obuf[k*16+a]; write x_out[a*256+m] directly
  gemm_bias_wmma<NB, 0, NB, 1, 0, 1, COUTc>
      <<<dim3(1, COUTc / 16), 32, 0, stream>>>(out_W2, out_b2, obuf, out_xout, COUTc, CFEATc);
}
